// Transformer_50156628082816
// MI455X (gfx1250) — matrix operand order repaired and checked by reading, not compiled, not run on hardware
//
#include <hip/hip_runtime.h>
#include <math.h>

// ---------------- model constants (match reference) ----------------
static constexpr int Bc  = 2;
static constexpr int Sq  = 1024;
static constexpr int Ed  = 1024;
static constexpr int NHd = 16;
static constexpr int AHd = 64;
static constexpr int HAd = NHd * AHd;   // 1024
static constexpr int NLd = 4;
static constexpr int HCd = 2048;
static constexpr int Vd  = 32000;
static constexpr int BSd = Bc * Sq;     // 2048

// ---------------- WMMA types ----------------
typedef __attribute__((ext_vector_type(16))) __bf16 v16bf;
typedef __attribute__((ext_vector_type(8)))  float  v8f;

#define BM 128
#define BN 128
#define BK 32
#define LDP 40   // padded LDS row stride (bf16 elems); 80B, multiple of 16B

#define GF_BT    1
#define GF_RELU  2
#define GF_BIAS  4
#define GF_OUTBF 8

__device__ __forceinline__ unsigned short f2bf(float f) {
    union { float f; unsigned u; } v; v.f = f;
    unsigned u = v.u;
    u += 0x7FFFu + ((u >> 16) & 1u);   // round-to-nearest-even
    return (unsigned short)(u >> 16);
}

union FragU { v16bf v; uint4 q[2]; };

__device__ __forceinline__ void async_cp_b128(unsigned lds_off, const void* gaddr) {
    // GLOBAL_LOAD_ASYNC_TO_LDS_B128 (GV mode): per-lane 16B global -> LDS, tracked by ASYNCcnt
    asm volatile("global_load_async_to_lds_b128 %0, %1, off"
                 :: "v"(lds_off), "v"((unsigned long long)(size_t)gaddr)
                 : "memory");
}
__device__ __forceinline__ void wait_async0() {
    asm volatile("s_wait_asynccnt 0" ::: "memory");
}

// ---------------- bf16 WMMA GEMM (bf16 in, f32 accumulate, f32/bf16 out) ----------------
// C[M,N] = act(alpha * A[M,K] @ B + bias), batched over (b,h).
// B layout: flags&GF_BT ? row-major [N,K] : row-major [K,N].
// Assumptions (true for every call below): M % 128 == 0, K % 32 == 0, N % 8 == 0,
// rows 16B-aligned. Only N may have partial tiles (N=64 in P@V).
__global__ __launch_bounds__(256)
void gemm_bf16_wmma(const unsigned short* __restrict__ A,
                    const unsigned short* __restrict__ B,
                    const float* __restrict__ bias, void* __restrict__ Cv,
                    int M, int N, int K, int lda, int ldb, int ldc,
                    int batH,
                    long long sAb, long long sAh,
                    long long sBb, long long sBh,
                    long long sCb, long long sCh,
                    float alpha, int flags)
{
    __shared__ unsigned short Als[2][BM * LDP];
    __shared__ unsigned short Bls[2][BN * LDP];

    const int tid = threadIdx.x;
    const int m0  = blockIdx.y * BM;
    const int n0  = blockIdx.x * BN;
    const int bz  = blockIdx.z;
    const int bb  = bz / batH;
    const int hh  = bz - bb * batH;
    A += bb * sAb + hh * sAh;
    B += bb * sBb + hh * sBh;
    const long long cOff = bb * sCb + hh * sCh;

    const int lane = tid & 31;
    const int wave = tid >> 5;
    const int wr   = wave >> 1;   // 0..3 -> 32-row strip
    const int wc   = wave & 1;    // 0..1 -> 64-col strip
    const int half = lane >> 4;   // 0/1
    const int lr   = lane & 15;

    const bool bT = (flags & GF_BT) != 0;

    // per-thread staging coordinates (k-invariant)
    const int stRow = (tid * 2) >> 2;        // 0..127 (two consecutive idx share row/seg+1)
    const int stSeg = (tid * 2) & 3;         // 16B segment (i=0); i=1 -> +1 (same row: 2*tid,2*tid+1 differ in seg only when seg<3)
    const int bnKk  = (tid * 2) >> 4;        // BN path: K row for i=0
    const int bnNs  = (tid * 2) & 15;        // BN path: 8-col segment

    v8f acc[2][4];
    for (int i = 0; i < 2; ++i)
        for (int j = 0; j < 4; ++j)
            for (int e = 0; e < 8; ++e) acc[i][j][e] = 0.0f;

    // row-major tile stage via async copies: 512 x 16B segments, 2 per thread
    auto stageMaj = [&](unsigned short* lbase, const unsigned short* gbase,
                        int ld, int off0, int k0) {
        for (int i = 0; i < 2; ++i) {
            const int idx = tid * 2 + i;
            const int row = idx >> 2;
            const int seg = idx & 3;
            async_cp_b128((unsigned)(size_t)(lbase + row * LDP + seg * 8),
                          gbase + (long long)(off0 + row) * ld + k0 + seg * 8);
        }
    };

    auto stageAll = [&](int buf, int k0) {
        stageMaj(&Als[buf][0], A, lda, m0, k0);
        if (bT) {
            stageMaj(&Bls[buf][0], B, ldb, n0, k0);   // always full tiles in our calls
        } else {
            // B [K,N] row-major: vector-load-or-zero (N % 8 == 0), scatter-transpose
            for (int i = 0; i < 2; ++i) {
                const int idx = tid * 2 + i;
                const int kk  = idx >> 4;             // 0..31
                const int ns  = idx & 15;             // 8-col segment
                const int gn  = n0 + ns * 8;
                uint4 q = {0u, 0u, 0u, 0u};
                if (gn + 7 < N)                        // k-invariant predicate
                    q = *(const uint4*)(B + (long long)(k0 + kk) * ldb + gn);
                unsigned short* d = &Bls[buf][ns * 8 * LDP + kk];
                d[0 * LDP] = (unsigned short)q.x; d[1 * LDP] = (unsigned short)(q.x >> 16);
                d[2 * LDP] = (unsigned short)q.y; d[3 * LDP] = (unsigned short)(q.y >> 16);
                d[4 * LDP] = (unsigned short)q.z; d[5 * LDP] = (unsigned short)(q.z >> 16);
                d[6 * LDP] = (unsigned short)q.w; d[7 * LDP] = (unsigned short)(q.w >> 16);
            }
        }
    };

    auto compute = [&](int buf) {
        FragU bf[4];
        for (int nt = 0; nt < 4; ++nt) {
            const int n = wc * 64 + nt * 16 + lr;
            const unsigned short* bp = &Bls[buf][n * LDP + half * 16];
            bf[nt].q[0] = *(const uint4*)(bp);
            bf[nt].q[1] = *(const uint4*)(bp + 8);
        }
        for (int mt = 0; mt < 2; ++mt) {
            const int row = wr * 32 + mt * 16 + lr;
            const unsigned short* ap = &Als[buf][row * LDP + half * 16];
            FragU af;
            af.q[0] = *(const uint4*)(ap);
            af.q[1] = *(const uint4*)(ap + 8);
            for (int nt = 0; nt < 4; ++nt) {
                acc[mt][nt] = __builtin_amdgcn_wmma_f32_16x16x32_bf16(
                    false, af.v, false, bf[nt].v, (short)0, acc[mt][nt], false, false);
            }
        }
    };

    const int ksteps = K / BK;
    stageAll(0, 0);                 // prologue: fill buffer 0
    wait_async0();
    __syncthreads();

    for (int ks = 0; ks < ksteps; ++ks) {
        const int cur = ks & 1;
        if (ks + 1 < ksteps) stageAll(cur ^ 1, (ks + 1) * BK);   // prefetch next
        compute(cur);
        wait_async0();        // own-wave prefetch landed
        __syncthreads();      // all waves' staging (async + ds) visible
    }
    (void)stRow; (void)stSeg; (void)bnKk; (void)bnNs;

    // ---- epilogue: alpha, bias, relu, store f32 or bf16 ----
    const bool outbf = (flags & GF_OUTBF) != 0;
    float* Cf = (float*)Cv;
    unsigned short* Cb = (unsigned short*)Cv;
    for (int mt = 0; mt < 2; ++mt) {
        for (int nt = 0; nt < 4; ++nt) {
            const int gn = n0 + wc * 64 + nt * 16 + lr;
            if (gn >= N) continue;
            const float bv = (flags & GF_BIAS) ? bias[gn] : 0.0f;
            for (int i = 0; i < 8; ++i) {
                const int gm = m0 + wr * 32 + mt * 16 + half * 8 + i;
                float v = alpha * acc[mt][nt][i] + bv;
                if (flags & GF_RELU) v = fmaxf(v, 0.0f);
                const long long ci = cOff + (long long)gm * ldc + gn;
                if (outbf) Cb[ci] = f2bf(v);
                else       Cf[ci] = v;
            }
        }
    }
}

// ---------------- f32 -> bf16 conversion (for weights) ----------------
__global__ __launch_bounds__(256)
void cvt_f32_bf16(const float* __restrict__ src, unsigned short* __restrict__ dst,
                  long long n)
{
    long long i = ((long long)blockIdx.x * 256 + threadIdx.x) * 8;
    if (i >= n) return;
    if (i + 7 < n) {
        const float4 a = *(const float4*)(src + i);
        const float4 b = *(const float4*)(src + i + 4);
        uint4 o;
        o.x = (unsigned)f2bf(a.x) | ((unsigned)f2bf(a.y) << 16);
        o.y = (unsigned)f2bf(a.z) | ((unsigned)f2bf(a.w) << 16);
        o.z = (unsigned)f2bf(b.x) | ((unsigned)f2bf(b.y) << 16);
        o.w = (unsigned)f2bf(b.z) | ((unsigned)f2bf(b.w) << 16);
        *(uint4*)(dst + i) = o;
    } else {
        for (; i < n; ++i) dst[i] = f2bf(src[i]);
    }
}

// ---------------- positional-encoding add (dual f32 + bf16 out) ----------------
__global__ __launch_bounds__(256)
void pe_add_kernel(const float* __restrict__ x, float* __restrict__ outf,
                   unsigned short* __restrict__ outb,
                   long long total, int S_, int E_)
{
    const long long idx = (long long)blockIdx.x * 256 + threadIdx.x;
    if (idx >= total) return;
    const int e = (int)(idx % E_);
    const int s = (int)((idx / E_) % S_);
    const float inv = (float)(e >> 1) / (float)E_;                 // (j//2)/E
    const float p  = (float)s * __expf(-9.210340371976184f * inv); // pos / 10000^inv
    const float pe = (e & 1) ? __cosf(p) : __sinf(p);
    const float v  = x[idx] + pe;
    outf[idx] = v;
    outb[idx] = f2bf(v);
}

// ---------------- fused residual-add + LayerNorm (row = E=1024) ----------------
// outf (f32) and/or outb (bf16) may be null. Contiguous 4-elem chunks per thread.
__global__ __launch_bounds__(256)
void add_ln_kernel(const float* __restrict__ a, const float* __restrict__ r,
                   const float* __restrict__ g, const float* __restrict__ b,
                   float* __restrict__ outf, unsigned short* __restrict__ outb,
                   int E_)
{
    __shared__ float red[256];
    const long long row = blockIdx.x;
    const float* pa = a + row * E_;
    const float* pr = r + row * E_;
    const int t = threadIdx.x;
    const int j0 = t * 4;

    const float4 va = *(const float4*)(pa + j0);
    const float4 vr = *(const float4*)(pr + j0);
    float v[4] = { va.x + vr.x, va.y + vr.y, va.z + vr.z, va.w + vr.w };
    float s = 0.f, s2 = 0.f;
    for (int i = 0; i < 4; ++i) { s += v[i]; s2 += v[i] * v[i]; }

    red[t] = s; __syncthreads();
    for (int o = 128; o > 0; o >>= 1) { if (t < o) red[t] += red[t + o]; __syncthreads(); }
    const float mean = red[0] / (float)E_;
    __syncthreads();
    red[t] = s2; __syncthreads();
    for (int o = 128; o > 0; o >>= 1) { if (t < o) red[t] += red[t + o]; __syncthreads(); }
    const float var  = red[0] / (float)E_ - mean * mean;
    const float rstd = rsqrtf(var + 1e-5f);

    const float4 vg = *(const float4*)(g + j0);
    const float4 vb = *(const float4*)(b + j0);
    float o0 = (v[0] - mean) * rstd * vg.x + vb.x;
    float o1 = (v[1] - mean) * rstd * vg.y + vb.y;
    float o2 = (v[2] - mean) * rstd * vg.z + vb.z;
    float o3 = (v[3] - mean) * rstd * vg.w + vb.w;
    if (outf) {
        float4 of = { o0, o1, o2, o3 };
        *(float4*)(outf + row * E_ + j0) = of;
    }
    if (outb) {
        uint2 ob;
        ob.x = (unsigned)f2bf(o0) | ((unsigned)f2bf(o1) << 16);
        ob.y = (unsigned)f2bf(o2) | ((unsigned)f2bf(o3) << 16);
        *(uint2*)(outb + row * E_ + j0) = ob;
    }
}

// ---------------- row softmax, f32 in -> bf16 probs out (row = S=1024) ----------------
__global__ __launch_bounds__(256)
void softmax_kernel(const float* __restrict__ sc, unsigned short* __restrict__ pbo,
                    int S_, int causal)
{
    __shared__ float red[256];
    const long long row = blockIdx.x;       // over B*H*S
    const int r = (int)(row % S_);
    const float* p = sc + row * S_;
    unsigned short* q = pbo + row * S_;
    const int t = threadIdx.x;
    const int j0 = t * 4;

    const float4 vx = *(const float4*)(p + j0);
    float v[4] = { vx.x, vx.y, vx.z, vx.w };
    float mx = -INFINITY;
    for (int i = 0; i < 4; ++i) {
        if (causal && (j0 + i) > r) v[i] = -INFINITY;
        mx = fmaxf(mx, v[i]);
    }
    red[t] = mx; __syncthreads();
    for (int o = 128; o > 0; o >>= 1) { if (t < o) red[t] = fmaxf(red[t], red[t + o]); __syncthreads(); }
    mx = red[0];
    __syncthreads();
    float sum = 0.f;
    for (int i = 0; i < 4; ++i) { const float e = __expf(v[i] - mx); v[i] = e; sum += e; }
    red[t] = sum; __syncthreads();
    for (int o = 128; o > 0; o >>= 1) { if (t < o) red[t] += red[t + o]; __syncthreads(); }
    const float inv = 1.0f / red[0];
    uint2 ob;
    ob.x = (unsigned)f2bf(v[0] * inv) | ((unsigned)f2bf(v[1] * inv) << 16);
    ob.y = (unsigned)f2bf(v[2] * inv) | ((unsigned)f2bf(v[3] * inv) << 16);
    *(uint2*)(q + j0) = ob;
}

// ---------------- host-side helpers ----------------
typedef unsigned short u16;

static void cvt(hipStream_t st, const float* s, u16* d, long long n) {
    const long long grp = (n + 8LL * 256 - 1) / (8LL * 256);
    cvt_f32_bf16<<<(int)grp, 256, 0, st>>>(s, d, n);
}

static void gemm(hipStream_t st, const u16* A, const u16* B, const float* bias, void* C,
                 int M, int N, int K, int lda, int ldb, int ldc,
                 int batB, int batH,
                 long long sAb, long long sAh, long long sBb, long long sBh,
                 long long sCb, long long sCh,
                 float alpha, int flags)
{
    dim3 g((N + BN - 1) / BN, (M + BM - 1) / BM, batB * batH);
    gemm_bf16_wmma<<<g, 256, 0, st>>>(A, B, bias, C, M, N, K, lda, ldb, ldc, batH,
                                      sAb, sAh, sBb, sBh, sCb, sCh, alpha, flags);
}

static void gemm_plain(hipStream_t st, const u16* A, const u16* B, const float* bias,
                       void* C, int M, int N, int K, float alpha, int flags)
{
    gemm(st, A, B, bias, C, M, N, K, K, N, N, 1, 1, 0, 0, 0, 0, 0, 0, alpha, flags);
}

// one MHA block: outf = softmax(alpha*(Qin Wq)(Kin Wk)^T + mask)(Vin Wv) Wo + bo
static void attention(hipStream_t st,
                      const u16* qin, const u16* kin, const u16* vin,
                      const float* wq, const float* wk, const float* wv,
                      const float* wo, const float* bo,
                      u16* wsm, u16* qb, u16* kb, u16* vb,
                      float* scores, u16* pb, u16* attnbb,
                      float* outf, int causal)
{
    const long long nEH = (long long)Ed * HAd;
    cvt(st, wq, wsm, nEH);
    gemm_plain(st, qin, wsm, nullptr, qb, BSd, HAd, Ed, 1.0f, GF_OUTBF);
    cvt(st, wk, wsm, nEH);
    gemm_plain(st, kin, wsm, nullptr, kb, BSd, HAd, Ed, 1.0f, GF_OUTBF);
    cvt(st, wv, wsm, nEH);
    gemm_plain(st, vin, wsm, nullptr, vb, BSd, HAd, Ed, 1.0f, GF_OUTBF);
    // scores[b,h] = (1/8) * Q K^T    (K is [Sk,AH] row-major == [N,K] layout -> GF_BT)
    gemm(st, qb, kb, nullptr, scores,
         Sq, Sq, AHd, HAd, HAd, Sq,
         Bc, NHd,
         (long long)Sq * HAd, (long long)AHd,
         (long long)Sq * HAd, (long long)AHd,
         (long long)NHd * Sq * Sq, (long long)Sq * Sq,
         0.125f, GF_BT);
    softmax_kernel<<<Bc * NHd * Sq, 256, 0, st>>>(scores, pb, Sq, causal);
    // attn[b,:,h,:] = P V
    gemm(st, pb, vb, nullptr, attnbb,
         Sq, AHd, Sq, Sq, HAd, HAd,
         Bc, NHd,
         (long long)NHd * Sq * Sq, (long long)Sq * Sq,
         (long long)Sq * HAd, (long long)AHd,
         (long long)Sq * HAd, (long long)AHd,
         1.0f, GF_OUTBF);
    cvt(st, wo, wsm, nEH);
    gemm_plain(st, attnbb, wsm, bo, outf, BSd, Ed, HAd, 1.0f, GF_BIAS);
}

extern "C" void kernel_launch(void* const* d_in, const int* in_sizes, int n_in,
                              void* d_out, int out_size, void* d_ws, size_t ws_size,
                              hipStream_t stream)
{
    (void)in_sizes; (void)n_in; (void)out_size; (void)ws_size;

    int p = 0;
    const float* x        = (const float*)d_in[p++];
    const float* y        = (const float*)d_in[p++];
    const float* enc_wq   = (const float*)d_in[p++];
    const float* enc_wk   = (const float*)d_in[p++];
    const float* enc_wv   = (const float*)d_in[p++];
    const float* enc_wo   = (const float*)d_in[p++];
    const float* enc_bo   = (const float*)d_in[p++];
    const float* enc_ln1g = (const float*)d_in[p++];
    const float* enc_ln1b = (const float*)d_in[p++];
    const float* enc_ln2g = (const float*)d_in[p++];
    const float* enc_ln2b = (const float*)d_in[p++];
    const float* enc_f1w  = (const float*)d_in[p++];
    const float* enc_f1b  = (const float*)d_in[p++];
    const float* enc_f2w  = (const float*)d_in[p++];
    const float* enc_f2b  = (const float*)d_in[p++];
    const float* dec_wq1  = (const float*)d_in[p++];
    const float* dec_wk1  = (const float*)d_in[p++];
    const float* dec_wv1  = (const float*)d_in[p++];
    const float* dec_wo1  = (const float*)d_in[p++];
    const float* dec_bo1  = (const float*)d_in[p++];
    const float* dec_wq2  = (const float*)d_in[p++];
    const float* dec_wk2  = (const float*)d_in[p++];
    const float* dec_wv2  = (const float*)d_in[p++];
    const float* dec_wo2  = (const float*)d_in[p++];
    const float* dec_bo2  = (const float*)d_in[p++];
    const float* dec_ln1g = (const float*)d_in[p++];
    const float* dec_ln1b = (const float*)d_in[p++];
    const float* dec_ln2g = (const float*)d_in[p++];
    const float* dec_ln2b = (const float*)d_in[p++];
    const float* dec_ln3g = (const float*)d_in[p++];
    const float* dec_ln3b = (const float*)d_in[p++];
    const float* dec_f1w  = (const float*)d_in[p++];
    const float* dec_f1b  = (const float*)d_in[p++];
    const float* dec_f2w  = (const float*)d_in[p++];
    const float* dec_f2b  = (const float*)d_in[p++];
    const float* cls_w1   = (const float*)d_in[p++];
    const float* cls_b1   = (const float*)d_in[p++];
    const float* cls_w2   = (const float*)d_in[p++];
    const float* cls_b2   = (const float*)d_in[p++];
    const float* cls_w3   = (const float*)d_in[p++];
    const float* cls_b3   = (const float*)d_in[p++];

    float* logits = (float*)d_out;

    // ---------------- workspace layout ----------------
    const size_t nBSE = (size_t)BSd * Ed;                  // 2,097,152
    const size_t nSS  = (size_t)Bc * NHd * Sq * Sq;        // 33,554,432

    char* wsb = (char*)d_ws;
    size_t o = 0;
    auto allocF = [&](size_t n) { float* r = (float*)(wsb + o); o += n * 4; return r; };
    auto allocH = [&](size_t n) { u16*   r = (u16*)  (wsb + o); o += n * 2; return r; };

    float* ex     = allocF(nBSE);     // encoder stream (f32 residual)
    float* dx     = allocF(nBSE);     // decoder stream
    float* t1     = allocF(nBSE);     // GEMM f32 out pre-LN
    float* scores = allocF(nSS);      // attention logits (f32)

    u16* exb    = allocH(nBSE);       // bf16 copies feeding WMMA
    u16* dxb    = allocH(nBSE);
    u16* qb     = allocH(nBSE);
    u16* kb     = allocH(nBSE);
    u16* vb     = allocH(nBSE);
    u16* attnbb = allocH(nBSE);
    u16* fxb    = allocH(nBSE);       // LN output feeding FF
    u16* ffhb   = allocH(nBSE);       // FF hidden (relu)
    u16* f1b    = allocH(nBSE);       // decoder f1 (query input of cross-attn)
    u16* pb     = allocH(nSS);        // softmax probs (bf16)
    u16* wsm    = allocH((size_t)HCd * HCd);  // weight scratch (largest layer/cls_w2)

    // cls_w3 (HC x V bf16 = 65.5M elems) reuses the scores region (33.5M f32) after attention
    u16* w3b = (u16*)scores;
    // classifier hiddens reuse pb after attention
    u16* h1b = pb;
    u16* h2b = pb + (size_t)BSd * HCd;

    const long long totalBSE = (long long)BSd * Ed;
    const int peBlocks = (int)((totalBSE + 255) / 256);

    // ---------------- encoder ----------------
    pe_add_kernel<<<peBlocks, 256, 0, stream>>>(x, ex, exb, totalBSE, Sq, Ed);
    for (int i = 0; i < NLd; ++i) {
        const long long wOff = (long long)i * Ed * HAd;
        const long long vOff = (long long)i * Ed;
        attention(stream, exb, exb, exb,
                  enc_wq + wOff, enc_wk + wOff, enc_wv + wOff,
                  enc_wo + wOff, enc_bo + vOff,
                  wsm, qb, kb, vb, scores, pb, attnbb, t1, /*causal=*/0);
        add_ln_kernel<<<BSd, 256, 0, stream>>>(t1, ex, enc_ln1g + vOff, enc_ln1b + vOff,
                                               nullptr, fxb, Ed);
        cvt(stream, enc_f1w + (long long)i * Ed * Ed, wsm, (long long)Ed * Ed);
        gemm_plain(stream, fxb, wsm, enc_f1b + vOff, ffhb, BSd, Ed, Ed, 1.0f,
                   GF_BIAS | GF_RELU | GF_OUTBF);
        cvt(stream, enc_f2w + (long long)i * Ed * Ed, wsm, (long long)Ed * Ed);
        gemm_plain(stream, ffhb, wsm, enc_f2b + vOff, t1, BSd, Ed, Ed, 1.0f, GF_BIAS);
        add_ln_kernel<<<BSd, 256, 0, stream>>>(t1, ex, enc_ln2g + vOff, enc_ln2b + vOff,
                                               ex, exb, Ed);
    }

    // ---------------- decoder ----------------
    pe_add_kernel<<<peBlocks, 256, 0, stream>>>(y, dx, dxb, totalBSE, Sq, Ed);
    for (int i = 0; i < NLd; ++i) {
        const long long wOff = (long long)i * Ed * HAd;
        const long long vOff = (long long)i * Ed;
        // masked self-attention
        attention(stream, dxb, dxb, dxb,
                  dec_wq1 + wOff, dec_wk1 + wOff, dec_wv1 + wOff,
                  dec_wo1 + wOff, dec_bo1 + vOff,
                  wsm, qb, kb, vb, scores, pb, attnbb, t1, /*causal=*/1);
        add_ln_kernel<<<BSd, 256, 0, stream>>>(t1, dx, dec_ln1g + vOff, dec_ln1b + vOff,
                                               nullptr, f1b, Ed);           // f1 (bf16 only)
        // cross-attention: Q from f1, K/V from encoder output
        attention(stream, f1b, exb, exb,
                  dec_wq2 + wOff, dec_wk2 + wOff, dec_wv2 + wOff,
                  dec_wo2 + wOff, dec_bo2 + vOff,
                  wsm, qb, kb, vb, scores, pb, attnbb, t1, /*causal=*/0);
        add_ln_kernel<<<BSd, 256, 0, stream>>>(t1, dx, dec_ln2g + vOff, dec_ln2b + vOff,
                                               nullptr, fxb, Ed);           // f2 (bf16 only)
        // feed-forward
        cvt(stream, dec_f1w + (long long)i * Ed * Ed, wsm, (long long)Ed * Ed);
        gemm_plain(stream, fxb, wsm, dec_f1b + vOff, ffhb, BSd, Ed, Ed, 1.0f,
                   GF_BIAS | GF_RELU | GF_OUTBF);
        cvt(stream, dec_f2w + (long long)i * Ed * Ed, wsm, (long long)Ed * Ed);
        gemm_plain(stream, ffhb, wsm, dec_f2b + vOff, t1, BSd, Ed, Ed, 1.0f, GF_BIAS);
        add_ln_kernel<<<BSd, 256, 0, stream>>>(t1, dx, dec_ln3g + vOff, dec_ln3b + vOff,
                                               dx, dxb, Ed);
    }

    // ---------------- classifier MLP ----------------
    cvt(stream, cls_w1, wsm, (long long)Ed * HCd);
    gemm_plain(stream, dxb, wsm, cls_b1, h1b, BSd, HCd, Ed, 1.0f,
               GF_BIAS | GF_RELU | GF_OUTBF);
    cvt(stream, cls_w2, wsm, (long long)HCd * HCd);
    gemm_plain(stream, h1b, wsm, cls_b2, h2b, BSd, HCd, HCd, 1.0f,
               GF_BIAS | GF_RELU | GF_OUTBF);
    cvt(stream, cls_w3, w3b, (long long)HCd * Vd);   // 131MB bf16: L2-resident
    gemm_plain(stream, h2b, w3b, cls_b3, logits, BSd, Vd, HCd, 1.0f, GF_BIAS);
}
